// analysisTransformModel_66297115181716
// MI455X (gfx1250) — compile-verified, matrix operand order checked by reading, not run on hardware
//
#include <hip/hip_runtime.h>

typedef __bf16 bf16_t;
typedef __attribute__((ext_vector_type(16))) __bf16 bf16x16;
typedef __attribute__((ext_vector_type(8)))  float   f32x8;

#define TILE_M 256
#define TILE_N 64
#define PADK   40   // 32 + 8 bf16 pad to dodge LDS bank conflicts

// ---------------------------------------------------------------------------
// WMMA fragment load from LDS (A row-major M x K, B supplied as N x K rows;
// layouts symmetric under transpose so the same loader serves both).
// Per CDNA5 16-bit A layout: lanes 0-15 -> K 0..7 & 16..23, lanes 16-31 ->
// K 8..15 & 24..31, row = lane & 15, pairs packed per VGPR.
// ---------------------------------------------------------------------------
__device__ inline bf16x16 load_frag(const bf16_t* base) {
  int lane = threadIdx.x & 31;
  int r  = lane & 15;
  int kb = (lane & 16) ? 8 : 0;
  const bf16_t* p = base + r * PADK + kb;
  union { unsigned u[8]; bf16x16 v; } f;
#pragma unroll
  for (int i = 0; i < 4; i++) f.u[i]     = *(const unsigned*)(p + 2 * i);
#pragma unroll
  for (int i = 0; i < 4; i++) f.u[4 + i] = *(const unsigned*)(p + 16 + 2 * i);
  return f.v;
}

// ---------------------------------------------------------------------------
// C[M,N] = A[M,K] * B[N,K]^T + bias (+gelu) (+resid); K % 32 == 0, N % 64 == 0.
// Block tile 256x64, 8 waves, each wave owns a 64x32 patch (4x2 WMMA tiles)
// -> 8 WMMAs per 6 fragment loads per k-step. EDGE=false path has zero
// per-iteration guards (host splits off the ragged M remainder).
// ---------------------------------------------------------------------------
template<bool EDGE>
__global__ __launch_bounds__(256)
void gemm_bf16_wmma(const bf16_t* __restrict__ A, const bf16_t* __restrict__ B,
                    const float* __restrict__ bias, const float* __restrict__ resid,
                    float* __restrict__ outF, bf16_t* __restrict__ outB,
                    int mBase, int M, int N, int K, int gelu)
{
  __shared__ __attribute__((aligned(16))) bf16_t sA[TILE_M * PADK]; // 20 KB
  __shared__ __attribute__((aligned(16))) bf16_t sB[TILE_N * PADK]; //  5 KB
  const int m0 = mBase + blockIdx.x * TILE_M;
  const int n0 = blockIdx.y * TILE_N;
  const int tid  = threadIdx.x;
  const int wave = tid >> 5;
  const int lane = tid & 31;
  const int wr = wave >> 1;   // 0..3 -> 64-row band
  const int wc = wave & 1;    // 0..1 -> 32-col band
  f32x8 acc[4][2] = {};

  for (int k0 = 0; k0 < K; k0 += 32) {
    { // Stage A: thread tid loads its whole 32-bf16 row slice (4x b128).
      int gr = m0 + tid;
      const bf16_t* src = A + (size_t)gr * K + k0;
      uint4 d0 = {0u,0u,0u,0u}, d1 = {0u,0u,0u,0u};
      uint4 d2 = {0u,0u,0u,0u}, d3 = {0u,0u,0u,0u};
      if (!EDGE || gr < M) {
        d0 = *(const uint4*)(src);
        d1 = *(const uint4*)(src + 8);
        d2 = *(const uint4*)(src + 16);
        d3 = *(const uint4*)(src + 24);
      }
      bf16_t* dst = sA + tid * PADK;
      *(uint4*)(dst)      = d0;
      *(uint4*)(dst + 8)  = d1;
      *(uint4*)(dst + 16) = d2;
      *(uint4*)(dst + 24) = d3;
    }
    { // Stage B: 64x32 bf16 = 256 chunks, 1 per thread. N%64==0 -> no guard.
      int row = tid >> 2;
      int kc  = (tid & 3) << 3;
      uint4 d = *(const uint4*)(B + (size_t)(n0 + row) * K + k0 + kc);
      *(uint4*)(sB + row * PADK + kc) = d;
    }
    __syncthreads();

    bf16x16 b0 = load_frag(sB + (wc * 32)      * PADK);
    bf16x16 b1 = load_frag(sB + (wc * 32 + 16) * PADK);
#pragma unroll
    for (int i = 0; i < 4; i++) {
      bf16x16 a = load_frag(sA + (wr * 64 + i * 16) * PADK);
      acc[i][0] = __builtin_amdgcn_wmma_f32_16x16x32_bf16(false, a, false, b0, (short)0, acc[i][0], false, false);
      acc[i][1] = __builtin_amdgcn_wmma_f32_16x16x32_bf16(false, a, false, b1, (short)0, acc[i][1], false, false);
    }
    __syncthreads();
  }

  // Epilogue: C/D layout -> lane&15 = N col, VGPR v = M row (+8 for hi lanes).
  const int ncol = lane & 15;
  const int moff = (lane & 16) ? 8 : 0;
#pragma unroll
  for (int i = 0; i < 4; i++)
#pragma unroll
    for (int j = 0; j < 2; j++) {
      int n = n0 + wc * 32 + j * 16 + ncol;
      float bv = bias ? bias[n] : 0.0f;
#pragma unroll
      for (int v = 0; v < 8; v++) {
        int m = m0 + wr * 64 + i * 16 + moff + v;
        if (EDGE && m >= M) continue;
        float val = acc[i][j][v] + bv;
        if (gelu) val = 0.5f * val * (1.0f + erff(val * 0.70710678f));
        size_t off = (size_t)m * N + n;
        if (resid) val += resid[off];
        if (outF) outF[off] = val;
        if (outB) outB[off] = (bf16_t)val;
      }
    }
}

// ---------------------------------------------------------------------------
// LayerNorm over last dim C, one 64-thread block per token. Optional f32 and
// bf16 outputs (bf16 feeds the WMMA GEMMs directly). Safe in-place for outF.
// ---------------------------------------------------------------------------
__global__ __launch_bounds__(64)
void ln_kernel(const float* __restrict__ x, const float* __restrict__ g,
               const float* __restrict__ b, float* outF, bf16_t* outB, int C)
{
  __shared__ float red[64];
  const long row = blockIdx.x;
  const float* xr = x + row * (long)C;
  float s = 0.0f;
  for (int c = threadIdx.x; c < C; c += 64) s += xr[c];
  red[threadIdx.x] = s; __syncthreads();
  for (int o = 32; o > 0; o >>= 1) { if (threadIdx.x < o) red[threadIdx.x] += red[threadIdx.x + o]; __syncthreads(); }
  float mean = red[0] / C; __syncthreads();
  float v = 0.0f;
  for (int c = threadIdx.x; c < C; c += 64) { float d = xr[c] - mean; v += d * d; }
  red[threadIdx.x] = v; __syncthreads();
  for (int o = 32; o > 0; o >>= 1) { if (threadIdx.x < o) red[threadIdx.x] += red[threadIdx.x + o]; __syncthreads(); }
  float rstd = rsqrtf(red[0] / C + 1e-5f);
  for (int c = threadIdx.x; c < C; c += 64) {
    float y = (xr[c] - mean) * rstd * g[c] + b[c];
    if (outF) outF[row * (long)C + c] = y;
    if (outB) outB[row * (long)C + c] = (bf16_t)y;
  }
}

// ---------------------------------------------------------------------------
// Window partition with pad + cyclic shift: (H*W, C) bf16 -> (nW*49, C) bf16.
// Padded positions are zero-filled (matches reference: pad after LN).
// ---------------------------------------------------------------------------
__global__ void win_part_kernel(const bf16_t* __restrict__ xn, bf16_t* __restrict__ xw,
                                int H, int W, int C, int Hp, int Wp, int shift)
{
  long i = blockIdx.x * (long)blockDim.x + threadIdx.x;
  int nWw = Wp / 7;
  long total = (long)(Hp / 7) * nWw * 49 * C;
  if (i >= total) return;
  int c = i % C; long r = i / C;
  int t = r % 49; long wi = r / 49;
  int wh = wi / nWw, ww = wi % nWw;
  int gh = wh * 7 + t / 7, gw = ww * 7 + t % 7;   // shifted-frame coords
  int sh = (gh + shift) % Hp, sw = (gw + shift) % Wp;
  bf16_t v = (bf16_t)0.0f;
  if (sh < H && sw < W) v = xn[((long)sh * W + sw) * C + c];
  xw[i] = v;
}

// ---------------------------------------------------------------------------
// Window reverse + residual add: x(H*W,C) += gather(proj(nW*49,C)).
// ---------------------------------------------------------------------------
__global__ void win_rev_add_kernel(const float* __restrict__ proj, float* __restrict__ x,
                                   int H, int W, int C, int Hp, int Wp, int shift)
{
  long i = blockIdx.x * (long)blockDim.x + threadIdx.x;
  long total = (long)H * W * C;
  if (i >= total) return;
  int c = i % C; long r = i / C;
  int h = r / W, w = r % W;
  int gi = (h - shift + Hp) % Hp, gj = (w - shift + Wp) % Wp;
  int nWw = Wp / 7;
  long wi = (long)(gi / 7) * nWw + (gj / 7);
  int t = (gi % 7) * 7 + (gj % 7);
  x[i] += proj[(wi * 49 + t) * (long)C + c];
}

// ---------------------------------------------------------------------------
// Window attention: one block per (window, head). N=49 is tiny (~4% of FLOPs)
// so VALU with LDS-staged q/k/v and LDS score rows (avoids scratch spills);
// rel-pos bias + shift mask computed analytically in-kernel.
// ---------------------------------------------------------------------------
__global__ __launch_bounds__(64)
void attn_kernel(const bf16_t* __restrict__ qkv, bf16_t* __restrict__ out,
                 const float* __restrict__ rpb,
                 int nH, int C, int Hp, int Wp, int shift)
{
  const int N = 49;
  int wi = blockIdx.x, h = blockIdx.y;
  int hd = C / nH;
  __shared__ bf16_t sq[49 * 96], sk[49 * 96], sv[49 * 96];
  __shared__ float sS[64 * 50];
  long rowbase = (long)wi * N;
  int total = N * hd;
  for (int i = threadIdx.x; i < total; i += 64) {
    int t = i / hd, d = i % hd;
    const bf16_t* r = qkv + (rowbase + t) * (long)(3 * C);
    sq[t * 96 + d] = r[h * hd + d];
    sk[t * 96 + d] = r[C + h * hd + d];
    sv[t * 96 + d] = r[2 * C + h * hd + d];
  }
  __syncthreads();
  int t = threadIdx.x;
  if (t >= N) return;
  float* s = sS + threadIdx.x * 50;
  float scale = rsqrtf((float)hd);
  int th = t / 7, tw = t % 7;
  int nWw = Wp / 7;
  int wh = wi / nWw, ww = wi % nWw;
  int reg_t = 0;
  if (shift > 0) {
    int gh = wh * 7 + th, gw = ww * 7 + tw;
    int rh = (gh < Hp - 7) ? 0 : (gh < Hp - shift ? 1 : 2);
    int rw = (gw < Wp - 7) ? 0 : (gw < Wp - shift ? 1 : 2);
    reg_t = rh * 3 + rw;
  }
  float mx = -1e30f;
  for (int j = 0; j < N; j++) {
    float acc = 0.0f;
    for (int d = 0; d < hd; d++)
      acc += (float)sq[t * 96 + d] * (float)sk[j * 96 + d];
    acc *= scale;
    int jh = j / 7, jw = j % 7;
    acc += rpb[((th - jh + 6) * 13 + (tw - jw + 6)) * nH + h];
    if (shift > 0) {
      int gjh = wh * 7 + jh, gjw = ww * 7 + jw;
      int rh = (gjh < Hp - 7) ? 0 : (gjh < Hp - shift ? 1 : 2);
      int rw = (gjw < Wp - 7) ? 0 : (gjw < Wp - shift ? 1 : 2);
      if (rh * 3 + rw != reg_t) acc -= 100.0f;
    }
    s[j] = acc;
    mx = fmaxf(mx, acc);
  }
  float sum = 0.0f;
  for (int j = 0; j < N; j++) { float e = __expf(s[j] - mx); s[j] = e; sum += e; }
  float inv = 1.0f / sum;
  for (int d = 0; d < hd; d++) {
    float o = 0.0f;
    for (int j = 0; j < N; j++) o += s[j] * (float)sv[j * 96 + d];
    out[(rowbase + t) * (long)C + h * hd + d] = (bf16_t)(o * inv);
  }
}

// ---------------------------------------------------------------------------
// Small utility kernels.
// ---------------------------------------------------------------------------
__global__ void cvt_bf16_kernel(const float* __restrict__ in, bf16_t* __restrict__ out, long n)
{ long i = blockIdx.x * (long)blockDim.x + threadIdx.x; if (i < n) out[i] = (bf16_t)in[i]; }

__global__ void add_kernel(float* __restrict__ d, const float* __restrict__ s, long n)
{ long i = blockIdx.x * (long)blockDim.x + threadIdx.x; if (i < n) d[i] += s[i]; }

__global__ void pack_w_kernel(const float* __restrict__ w, bf16_t* __restrict__ out,
                              int N, int K, int Kp)
{
  long i = blockIdx.x * (long)blockDim.x + threadIdx.x;
  if (i >= (long)N * Kp) return;
  int n = i / Kp, k = i % Kp;
  out[i] = (k < K) ? (bf16_t)w[(long)n * K + k] : (bf16_t)0.0f;
}

// im2col for 5x5 stride-2 pad-2 convs; layout 0 = NCHW input, 1 = (HW,C) tokens.
__global__ void im2col_kernel(const float* __restrict__ x, bf16_t* __restrict__ out,
                              int layout, int H, int W, int Cin, int Ho, int Wo, int Kp)
{
  long i = blockIdx.x * (long)blockDim.x + threadIdx.x;
  long total = (long)Ho * Wo * Kp;
  if (i >= total) return;
  int k = i % Kp; long row = i / Kp;
  float v = 0.0f;
  if (k < Cin * 25) {
    int cin = k / 25, r = k % 25, kh = r / 5, kw = r % 5;
    int oh = row / Wo, ow = row % Wo;
    int ih = oh * 2 - 2 + kh, iw = ow * 2 - 2 + kw;
    if (ih >= 0 && ih < H && iw >= 0 && iw < W)
      v = (layout == 0) ? x[((long)cin * H + ih) * W + iw]
                        : x[((long)ih * W + iw) * Cin + cin];
  }
  out[i] = (bf16_t)v;
}

__global__ void tok_to_nchw_kernel(const float* __restrict__ t, float* __restrict__ out,
                                   int HW, int Cn)
{
  long i = blockIdx.x * (long)blockDim.x + threadIdx.x;
  if (i >= (long)HW * Cn) return;
  int hw = i / Cn, c = i % Cn;
  out[(long)c * HW + hw] = t[i];
}

// ---------------------------------------------------------------------------
// Host-side orchestration.
// ---------------------------------------------------------------------------
struct Work {
  float  *T0, *T1, *XCUR, *PROJ;
  bf16_t *LN, *XW, *AO, *WB, *QKV, *HID, *PATCH;
};

static inline const float* FP(void* const* d_in, int i) { return (const float*)d_in[i]; }

static void launch_gemm(const bf16_t* A, const bf16_t* B, const float* bias,
                        const float* resid, float* outF, bf16_t* outB,
                        int M, int N, int K, int gelu, hipStream_t s)
{
  int mFull = (M / TILE_M) * TILE_M;
  if (mFull > 0) {
    dim3 g(mFull / TILE_M, N / TILE_N);
    gemm_bf16_wmma<false><<<g, 256, 0, s>>>(A, B, bias, resid, outF, outB, 0, M, N, K, gelu);
  }
  if (mFull < M) {
    dim3 g(1, N / TILE_N);
    gemm_bf16_wmma<true><<<g, 256, 0, s>>>(A, B, bias, resid, outF, outB, mFull, M, N, K, gelu);
  }
}

static void launch_pack(const float* w, bf16_t* out, int N, int K, int Kp, hipStream_t s)
{
  long tot = (long)N * Kp;
  pack_w_kernel<<<(unsigned)((tot + 255) / 256), 256, 0, s>>>(w, out, N, K, Kp);
}

static void conv5x5(const float* in, int layout, int H, int W, int Cin, int Cout,
                    const float* wgt, const float* bias, float* out,
                    const Work& wk, hipStream_t s)
{
  int Ho = H / 2, Wo = W / 2;           // k=5, stride=2, pad=2, even H/W
  int K = Cin * 25, Kp = (K + 31) / 32 * 32;
  long tot = (long)Ho * Wo * Kp;
  im2col_kernel<<<(unsigned)((tot + 255) / 256), 256, 0, s>>>(in, wk.PATCH, layout, H, W, Cin, Ho, Wo, Kp);
  launch_pack(wgt, wk.WB, Cout, K, Kp, s);
  launch_gemm(wk.PATCH, wk.WB, bias, nullptr, out, nullptr, Ho * Wo, Cout, Kp, 0, s);
}

static void swin_stage(void* const* d_in, int base, int nblk,
                       const int* nhs, const int* shifts,
                       float* xbase, int H, int W, int C,
                       const Work& wk, hipStream_t s)
{
  const long L = (long)H * W;
  const int Hp = H + (7 - H % 7) % 7, Wp = W + (7 - W % 7) % 7;
  const int nW = (Hp / 7) * (Wp / 7);
  const long LW = (long)nW * 49;

  // patch embed (1x1 conv == GEMM) + embed LN -> XCUR (the running x)
  cvt_bf16_kernel<<<(unsigned)((L * C + 255) / 256), 256, 0, s>>>(xbase, wk.LN, L * C);
  launch_pack(FP(d_in, base + 0), wk.WB, C, C, C, s);
  launch_gemm(wk.LN, wk.WB, FP(d_in, base + 1), nullptr, wk.XCUR, nullptr, (int)L, C, C, 0, s);
  ln_kernel<<<(unsigned)L, 64, 0, s>>>(wk.XCUR, FP(d_in, base + 2), FP(d_in, base + 3), wk.XCUR, nullptr, C);

  int p = base + 4;
  for (int i = 0; i < nblk; i++, p += 13) {
    const int sh = shifts[i], nH = nhs[i];
    // LN1 -> bf16
    ln_kernel<<<(unsigned)L, 64, 0, s>>>(wk.XCUR, FP(d_in, p + 0), FP(d_in, p + 1), nullptr, wk.LN, C);
    // window partition (pad + shift)
    long wtot = LW * C;
    win_part_kernel<<<(unsigned)((wtot + 255) / 256), 256, 0, s>>>(wk.LN, wk.XW, H, W, C, Hp, Wp, sh);
    // qkv GEMM
    launch_pack(FP(d_in, p + 2), wk.WB, 3 * C, C, C, s);
    launch_gemm(wk.XW, wk.WB, FP(d_in, p + 3), nullptr, nullptr, wk.QKV, (int)LW, 3 * C, C, 0, s);
    // windowed attention
    attn_kernel<<<dim3(nW, nH), 64, 0, s>>>(wk.QKV, wk.AO, FP(d_in, p + 4), nH, C, Hp, Wp, sh);
    // proj GEMM -> fp32
    launch_pack(FP(d_in, p + 5), wk.WB, C, C, C, s);
    launch_gemm(wk.AO, wk.WB, FP(d_in, p + 6), nullptr, wk.PROJ, nullptr, (int)LW, C, C, 0, s);
    // window reverse + residual
    long rtot = L * C;
    win_rev_add_kernel<<<(unsigned)((rtot + 255) / 256), 256, 0, s>>>(wk.PROJ, wk.XCUR, H, W, C, Hp, Wp, sh);
    // MLP: LN2 -> fc1(GELU) -> fc2(+residual)
    ln_kernel<<<(unsigned)L, 64, 0, s>>>(wk.XCUR, FP(d_in, p + 7), FP(d_in, p + 8), nullptr, wk.LN, C);
    launch_pack(FP(d_in, p + 9), wk.WB, 4 * C, C, C, s);
    launch_gemm(wk.LN, wk.WB, FP(d_in, p + 10), nullptr, nullptr, wk.HID, (int)L, 4 * C, C, 1, s);
    launch_pack(FP(d_in, p + 11), wk.WB, C, 4 * C, 4 * C, s);
    launch_gemm(wk.HID, wk.WB, FP(d_in, p + 12), wk.XCUR, wk.XCUR, nullptr, (int)L, C, 4 * C, 0, s);
  }
  // final norm + outer residual: xbase += norm(XCUR)
  ln_kernel<<<(unsigned)L, 64, 0, s>>>(wk.XCUR, FP(d_in, p + 0), FP(d_in, p + 1), wk.PROJ, nullptr, C);
  add_kernel<<<(unsigned)((L * C + 255) / 256), 256, 0, s>>>(xbase, wk.PROJ, L * C);
}

extern "C" void kernel_launch(void* const* d_in, const int* in_sizes, int n_in,
                              void* d_out, int out_size, void* d_ws, size_t ws_size,
                              hipStream_t stream)
{
  (void)in_sizes; (void)n_in; (void)out_size; (void)ws_size;
  const int C = 192;
  const long L1  = 65536;          // 256*256 tokens, stage 1
  const long LW1 = 1369L * 49;     // 37x37 windows of 49

  char* base = (char*)d_ws;
  size_t off = 0;
  auto alloc = [&](size_t bytes) -> void* {
    void* p = base + off;
    off = (off + bytes + 255) & ~(size_t)255;
    return p;
  };

  Work wk;
  wk.T0   = (float*)alloc((size_t)L1 * C * 4);
  wk.T1   = (float*)alloc((size_t)L1 * C * 4);
  wk.XCUR = (float*)alloc((size_t)L1 * C * 4);
  wk.PROJ = (float*)alloc((size_t)LW1 * C * 4);
  wk.LN   = (bf16_t*)alloc((size_t)L1 * C * 2);
  wk.XW   = (bf16_t*)alloc((size_t)LW1 * C * 2);
  wk.AO   = (bf16_t*)alloc((size_t)LW1 * C * 2);
  wk.WB   = (bf16_t*)alloc((size_t)384 * 4800 * 2);
  // QKV + HID contiguous; im2col PATCH (157 MB max) overlays them (never live
  // at the same time as the attention/MLP phase buffers).
  size_t qkv_b = (size_t)LW1 * 3 * C * 2;
  size_t hid_b = (size_t)L1 * 4 * C * 2;
  char* big = (char*)alloc(qkv_b + hid_b);
  wk.QKV   = (bf16_t*)big;
  wk.HID   = (bf16_t*)(big + qkv_b);
  wk.PATCH = (bf16_t*)big;

  // ---- parameter index map (setup_inputs dict order, depth-first) ----
  // 0: inputs; 1..8: c1w,c1b,c2w,c2b,c3w,c3b,c4w,c4b
  // 9: s1.embed_w ... (s1 blocks: 13 tensors each, 12 blocks), s1 norm
  const int S1 = 9;
  const int S3 = S1 + 4 + 12 * 13 + 2;   // == 171

  static const int nhs1[12] = {2,2, 2,2, 6,6,6,6,6,6, 2,2};
  static const int sh1 [12] = {0,3, 0,3, 0,3,0,3,0,3, 0,3};
  static const int nhs3[4]  = {2,2, 2,2};
  static const int sh3 [4]  = {0,3, 0,3};

  // conv1: NCHW (3,512,512) -> tokens (65536,192)
  conv5x5(FP(d_in, 0), 0, 512, 512, 3, 192, FP(d_in, 1), FP(d_in, 2), wk.T0, wk, stream);
  // swin stage 1 (residual added into T0)
  swin_stage(d_in, S1, 12, nhs1, sh1, wk.T0, 256, 256, C, wk, stream);
  // conv2 -> (16384,192); swin2 residual is dead code in the reference
  conv5x5(wk.T0, 1, 256, 256, 192, 192, FP(d_in, 3), FP(d_in, 4), wk.T1, wk, stream);
  // conv3 -> (4096,192)
  conv5x5(wk.T1, 1, 128, 128, 192, 192, FP(d_in, 5), FP(d_in, 6), wk.T0, wk, stream);
  // swin stage 3
  swin_stage(d_in, S3, 4, nhs3, sh3, wk.T0, 64, 64, C, wk, stream);
  // conv4 -> (1024,384); swin4 residual is dead code
  conv5x5(wk.T0, 1, 64, 64, 192, 384, FP(d_in, 7), FP(d_in, 8), wk.T1, wk, stream);
  // tokens (HW,C) -> NCHW output
  long tot = 1024L * 384;
  tok_to_nchw_kernel<<<(unsigned)((tot + 255) / 256), 256, 0, stream>>>(wk.T1, (float*)d_out, 1024, 384);
}
